// MultiHeadAttentionBlock_40733469835641
// MI455X (gfx1250) — matrix operand order repaired and checked by reading, not compiled, not run on hardware
//
#include <hip/hip_runtime.h>
#include <cstdint>

// ---------------------------------------------------------------------------
// MultiHeadAttention for MI455X (gfx1250, wave32, WMMA f32_16x16x32_f16,
// Tensor Data Mover staging of K/V tiles into LDS shared by 4 waves).
// B=4, S=2048, D_MODEL=1024, H=16, D_K=64
// Workspace layout (assumed >= 73 MB):
//   [ 0MB) Qb   f16 [b,h,s,d]   16MB
//   [16MB) Kb   f16 [b,h,s,d]   16MB
//   [32MB) VTb  f16 [b,h,d,s]   16MB
//   [48MB) AO   f16 [b*s, 1024] 16MB (merged heads attention output)
//   [64MB) W*h  f16 4 x 2MB     8MB  (Wq,Wk,Wv,Wo converted)
//   [72MB) mbits u32            512KB (mask packed to 1 bit/element)
// ---------------------------------------------------------------------------

#define B_  4
#define S_  2048
#define D_  1024
#define H_  16
#define DK_ 64
#define WAVES_ 4   // waves per attention workgroup sharing K/V LDS tiles

typedef __attribute__((ext_vector_type(16))) _Float16 v16h;
typedef __attribute__((ext_vector_type(8)))  float    v8f;
typedef __attribute__((ext_vector_type(4)))  unsigned int u32x4;
typedef __attribute__((ext_vector_type(8)))  int      i32x8;
typedef __attribute__((ext_vector_type(4)))  int      i32x4;

union AFrag { v16h v; uint32_t u[8]; };

#if defined(__HIP_DEVICE_COMPILE__) && \
    __has_builtin(__builtin_amdgcn_tensor_load_to_lds) && \
    __has_builtin(__builtin_amdgcn_s_wait_tensorcnt)
#define HAVE_TDM 1
#else
#define HAVE_TDM 0
#endif

__device__ __forceinline__ uint32_t pack2h(float a, float b) {
  union { _Float16 h[2]; uint32_t u; } x;
  x.h[0] = (_Float16)a; x.h[1] = (_Float16)b;
  return x.u;
}
__device__ __forceinline__ unsigned short h1bits(float a) {
  union { _Float16 h; unsigned short u; } x;
  x.h = (_Float16)a; return x.u;
}

#if HAVE_TDM
// Build TDM D# groups 0/1 for a 2-D f16 tile load (see CDNA5 ISA D# layout).
//  tile: tile0 (contiguous elems) x tile1 (rows); row stride = stride0 elems.
__device__ __forceinline__ void tdm_load_2d_f16(
    const void* gptr, uint32_t lds_byte_addr,
    uint32_t tensor0, uint32_t tensor1, uint32_t tile0, uint32_t tile1,
    uint64_t stride0) {
  const uint64_t ga = (uint64_t)(uintptr_t)gptr;
  u32x4 g0;
  g0[0] = 1u;                                   // count=1, user desc, no gather
  g0[1] = lds_byte_addr;                        // lds_addr (bytes)
  g0[2] = (uint32_t)ga;                         // global_addr[31:0]
  g0[3] = (uint32_t)((ga >> 32) & 0x01FFFFFFu)  // global_addr[56:32]
          | 0x80000000u;                        // type=2 ("image")
  i32x8 g1;
  g1[0] = (int)(1u << 16);                      // data_size=1 (2 bytes)
  g1[1] = (int)((tensor0 & 0xFFFFu) << 16);     // tensor_dim0[15:0] @bits63:48
  g1[2] = (int)(((tensor0 >> 16) & 0xFFFFu) |   // tensor_dim0[31:16]
                ((tensor1 & 0xFFFFu) << 16));   // tensor_dim1[15:0]
  g1[3] = (int)(((tensor1 >> 16) & 0xFFFFu) |   // tensor_dim1[31:16]
                (tile0 << 16));                 // tile_dim0 @bits127:112
  g1[4] = (int)(tile1 & 0xFFFFu);               // tile_dim1 (tile_dim2=0)
  g1[5] = (int)(uint32_t)stride0;               // tensor_dim0_stride[31:0]
  g1[6] = (int)(uint32_t)((stride0 >> 32) & 0xFFFFu);
  g1[7] = 0;
  i32x4 z4 = {0, 0, 0, 0};
#if __clang_major__ >= 23
  i32x8 z8 = {0, 0, 0, 0, 0, 0, 0, 0};
  __builtin_amdgcn_tensor_load_to_lds(g0, g1, z4, z4, z8, 0);
#else
  __builtin_amdgcn_tensor_load_to_lds(g0, g1, z4, z4, 0);
#endif
}
#endif

// --------------------------- fp32 -> f16 convert ---------------------------
__global__ void cvt_f32_f16_kernel(const float* __restrict__ src,
                                   _Float16* __restrict__ dst, int n4) {
  int i = (blockIdx.x * blockDim.x + threadIdx.x) * 4;
  if (i < n4 * 4) {
    float4 f = *(const float4*)(src + i);
    dst[i + 0] = (_Float16)f.x;
    dst[i + 1] = (_Float16)f.y;
    dst[i + 2] = (_Float16)f.z;
    dst[i + 3] = (_Float16)f.w;
  }
}

// --------------------------- mask -> bitmask -------------------------------
__global__ void packmask_kernel(const int* __restrict__ mask,
                                uint32_t* __restrict__ bits) {
  int w = blockIdx.x * blockDim.x + threadIdx.x;   // [0, S*S/32)
  const int* p = mask + (size_t)w * 32;
  uint32_t v = 0;
  #pragma unroll
  for (int j = 0; j < 32; ++j) v |= (p[j] != 0 ? 1u : 0u) << j;
  bits[w] = v;
}

// --------------------- QKV projection: Y = X @ W^T + b ---------------------
__global__ __launch_bounds__(32)
void proj_kernel(const float* __restrict__ X, const _Float16* __restrict__ Wh,
                 const float* __restrict__ bias, unsigned short* __restrict__ out,
                 int vt_layout) {
  const int lane = threadIdx.x;
  const int n  = lane & 15;
  const int hi = lane >> 4;
  const int rowbase = blockIdx.x * 16;   // [0, 8192)
  const int colbase = blockIdx.y * 64;   // [0, 1024)

  v8f acc0 = {}, acc1 = {}, acc2 = {}, acc3 = {};

  for (int kb = 0; kb < D_; kb += 32) {
    AFrag a;
    const float* xrow = X + (size_t)(rowbase + n) * D_ + kb;
    #pragma unroll
    for (int j = 0; j < 4; ++j) {
      int k0 = hi * 8 + 2 * j;
      int k1 = 16 + hi * 8 + 2 * j;
      float2 f0 = *(const float2*)(xrow + k0);
      float2 f1 = *(const float2*)(xrow + k1);
      a.u[j]     = pack2h(f0.x, f0.y);
      a.u[j + 4] = pack2h(f1.x, f1.y);
    }
    #pragma unroll
    for (int t = 0; t < 4; ++t) {
      AFrag bf;
      const uint32_t* wrow =
          (const uint32_t*)(Wh + (size_t)(colbase + t * 16 + n) * D_ + kb);
      #pragma unroll
      for (int j = 0; j < 8; ++j) bf.u[j] = wrow[(j >> 2) * 8 + hi * 4 + (j & 3)];
      v8f* ac = (t == 0) ? &acc0 : (t == 1) ? &acc1 : (t == 2) ? &acc2 : &acc3;
      *ac = __builtin_amdgcn_wmma_f32_16x16x32_f16(false, a.v, false, bf.v,
                                                   (short)0, *ac, false, false);
    }
  }

  #pragma unroll
  for (int t = 0; t < 4; ++t) {
    const v8f ac = (t == 0) ? acc0 : (t == 1) ? acc1 : (t == 2) ? acc2 : acc3;
    const int col = colbase + t * 16 + n;
    const float bv = bias[col];
    const int hh = col >> 6, d = col & 63;
    #pragma unroll
    for (int r = 0; r < 8; ++r) {
      const int row = rowbase + r + 8 * hi;
      const int b = row >> 11, s = row & (S_ - 1);
      size_t off;
      if (vt_layout)
        off = (((size_t)(b * H_ + hh)) * DK_ + d) * S_ + s;
      else
        off = (((size_t)(b * H_ + hh)) * S_ + s) * DK_ + d;
      out[off] = h1bits(ac[r] + bv);
    }
  }
}

// ------------------------- flash attention per head ------------------------
// grid = (S/64, H, B); 4 waves/block. Each wave owns a 16-row q tile; the
// block shares K (32x64) and V (64x32) tiles staged in LDS by the TDM.
__global__ __launch_bounds__(32 * WAVES_)
void attn_kernel(const _Float16* __restrict__ Qb, const _Float16* __restrict__ Kb,
                 const _Float16* __restrict__ VTb, const uint32_t* __restrict__ mbits,
                 _Float16* __restrict__ AO) {
  const int tid  = threadIdx.x;
  const int lane = tid & 31;
  const int wid  = tid >> 5;
  const int n  = lane & 15;
  const int hi = lane >> 4;
  const int h = blockIdx.y, b = blockIdx.z;
  const int qbase = blockIdx.x * (16 * WAVES_) + wid * 16;
  const size_t headoff = (size_t)(b * H_ + h) * S_ * DK_;
  const _Float16* Qh = Qb + headoff;    // [s][d]
  const _Float16* Kh = Kb + headoff;    // [s][d]
  const _Float16* Vt = VTb + headoff;   // [d][s]

  __shared__ unsigned short Klds[32 * DK_];          // [key][d]   4KB
  __shared__ unsigned short Vlds[DK_ * 32];          // [d][s]     4KB
  __shared__ unsigned short pbuf[WAVES_][16 * 32];   // P staging  4KB

  // Q A-fragments (persist): 16x32 over d=0..31 and d=32..63
  AFrag qa0, qa1;
  {
    const uint32_t* qrow = (const uint32_t*)(Qh + (size_t)(qbase + n) * DK_);
    #pragma unroll
    for (int j = 0; j < 4; ++j) {
      int k0 = hi * 8 + 2 * j;
      int k1 = 16 + hi * 8 + 2 * j;
      qa0.u[j]     = qrow[k0 >> 1];
      qa0.u[j + 4] = qrow[k1 >> 1];
      qa1.u[j]     = qrow[(32 + k0) >> 1];
      qa1.u[j + 4] = qrow[(32 + k1) >> 1];
    }
  }

  v8f o0 = {}, o1 = {}, o2 = {}, o3 = {};
  float rowM[8], rowL[8];
  #pragma unroll
  for (int r = 0; r < 8; ++r) { rowM[r] = -3.0e38f; rowL[r] = 0.0f; }

  for (int kk = 0; kk < S_ / 32; ++kk) {
    const int kbase = kk * 32;

    __syncthreads();   // previous tiles fully consumed
#if HAVE_TDM
    if (wid == 0) {    // wave 0 issues both TDM descriptors (EXEC ignored)
      // K tile: 32 rows x 64 f16, contiguous rows (stride 64 elems)
      tdm_load_2d_f16(Kh + (size_t)kbase * DK_,
                      (uint32_t)(uintptr_t)&Klds[0],
                      DK_, 32, DK_, 32, DK_);
      // V tile: 64 rows x 32 f16, row stride S (strided: TDM's sweet spot)
      tdm_load_2d_f16(Vt + kbase,
                      (uint32_t)(uintptr_t)&Vlds[0],
                      32, DK_, 32, DK_, S_);
      __builtin_amdgcn_s_wait_tensorcnt(0);
    }
#else
    {  // cooperative fallback: 128 threads copy 8KB
      const uint4* ks = (const uint4*)(Kh + (size_t)kbase * DK_);
      uint4* kd = (uint4*)&Klds[0];
      kd[tid]       = ks[tid];
      kd[tid + 128] = ks[tid + 128];
      uint4* vd = (uint4*)&Vlds[0];
      #pragma unroll
      for (int c = 0; c < 2; ++c) {
        int chunk = tid + c * 128;             // 256 chunks of 16B
        int row = chunk >> 2, col = chunk & 3;
        vd[chunk] = *(const uint4*)(Vt + (size_t)row * S_ + kbase + col * 8);
      }
    }
#endif
    __syncthreads();   // staged tiles visible to all 4 waves

    // prefetch next key tile toward L2 (global_prefetch_b8)
    if (kk + 1 < S_ / 32)
      __builtin_prefetch(Kh + (size_t)(kbase + 32 + lane) * DK_, 0, 3);

    // K B-fragments from LDS: B[k=d][n=key] = K[key][d]
    AFrag kb00, kb01, kb10, kb11;
    {
      const uint32_t* kr0 = (const uint32_t*)&Klds[(size_t)n * DK_];
      const uint32_t* kr1 = (const uint32_t*)&Klds[(size_t)(16 + n) * DK_];
      #pragma unroll
      for (int j = 0; j < 8; ++j) {
        kb00.u[j] = kr0[(j >> 2) * 8 + hi * 4 + (j & 3)];
        kb01.u[j] = kr0[16 + (j >> 2) * 8 + hi * 4 + (j & 3)];
        kb10.u[j] = kr1[(j >> 2) * 8 + hi * 4 + (j & 3)];
        kb11.u[j] = kr1[16 + (j >> 2) * 8 + hi * 4 + (j & 3)];
      }
    }
    v8f s0 = {}, s1 = {};
    s0 = __builtin_amdgcn_wmma_f32_16x16x32_f16(false, qa0.v, false, kb00.v,
                                                (short)0, s0, false, false);
    s0 = __builtin_amdgcn_wmma_f32_16x16x32_f16(false, qa1.v, false, kb01.v,
                                                (short)0, s0, false, false);
    s1 = __builtin_amdgcn_wmma_f32_16x16x32_f16(false, qa0.v, false, kb10.v,
                                                (short)0, s1, false, false);
    s1 = __builtin_amdgcn_wmma_f32_16x16x32_f16(false, qa1.v, false, kb11.v,
                                                (short)0, s1, false, false);

    uint32_t mw[8];
    #pragma unroll
    for (int r = 0; r < 8; ++r)
      mw[r] = mbits[(size_t)(qbase + r + 8 * hi) * (S_ / 32) + kk];

    // online softmax; half-wave shfl reductions match C-fragment row ownership
    float p0[8], p1[8];
    #pragma unroll
    for (int r = 0; r < 8; ++r) {
      float a0 = ((mw[r] >> n) & 1u)        ? s0[r] * 0.125f : -1.0e9f;
      float a1 = ((mw[r] >> (16 + n)) & 1u) ? s1[r] * 0.125f : -1.0e9f;
      float bm = fmaxf(a0, a1);
      #pragma unroll
      for (int off = 1; off < 16; off <<= 1)
        bm = fmaxf(bm, __shfl_xor(bm, off, 32));
      float newM  = fmaxf(rowM[r], bm);
      float alpha = __expf(rowM[r] - newM);
      rowM[r] = newM;
      float e0 = __expf(a0 - newM);
      float e1 = __expf(a1 - newM);
      float rs = e0 + e1;
      #pragma unroll
      for (int off = 1; off < 16; off <<= 1) rs += __shfl_xor(rs, off, 32);
      rowL[r] = rowL[r] * alpha + rs;
      o0[r] *= alpha; o1[r] *= alpha; o2[r] *= alpha; o3[r] *= alpha;
      p0[r] = e0; p1[r] = e1;
    }

    // P: C-layout -> A-layout via per-wave LDS tile (DS in-order per wave)
    #pragma unroll
    for (int r = 0; r < 8; ++r) {
      const int row = r + 8 * hi;
      pbuf[wid][row * 32 + n]      = h1bits(p0[r]);
      pbuf[wid][row * 32 + 16 + n] = h1bits(p1[r]);
    }
    AFrag pa;
    #pragma unroll
    for (int j = 0; j < 4; ++j) {
      int k0 = hi * 8 + 2 * j;
      int k1 = 16 + hi * 8 + 2 * j;
      pa.u[j]     = *(const uint32_t*)&pbuf[wid][n * 32 + k0];
      pa.u[j + 4] = *(const uint32_t*)&pbuf[wid][n * 32 + k1];
    }

    // O += P @ V from LDS: B[k=s][n=d] = Vlds[d][s]
    #pragma unroll
    for (int t = 0; t < 4; ++t) {
      AFrag vb;
      const uint32_t* vrow = (const uint32_t*)&Vlds[(size_t)(t * 16 + n) * 32];
      #pragma unroll
      for (int j = 0; j < 8; ++j) vb.u[j] = vrow[(j >> 2) * 8 + hi * 4 + (j & 3)];
      v8f* oc = (t == 0) ? &o0 : (t == 1) ? &o1 : (t == 2) ? &o2 : &o3;
      *oc = __builtin_amdgcn_wmma_f32_16x16x32_f16(false, pa.v, false, vb.v,
                                                   (short)0, *oc, false, false);
    }
  }

  // normalize and store merged-head f16 [b*s, 1024]
  #pragma unroll
  for (int r = 0; r < 8; ++r) {
    const float inv = 1.0f / rowL[r];
    const int srow = qbase + r + 8 * hi;
    const size_t base = ((size_t)b * S_ + srow) * D_ + h * DK_;
    AO[base +  0 + n] = (_Float16)(o0[r] * inv);
    AO[base + 16 + n] = (_Float16)(o1[r] * inv);
    AO[base + 32 + n] = (_Float16)(o2[r] * inv);
    AO[base + 48 + n] = (_Float16)(o3[r] * inv);
  }
}

// ------------------- output projection: out = AO @ Wo^T + bo ---------------
__global__ __launch_bounds__(32)
void oproj_kernel(const _Float16* __restrict__ A, const _Float16* __restrict__ Wh,
                  const float* __restrict__ bias, float* __restrict__ out) {
  const int lane = threadIdx.x;
  const int n  = lane & 15;
  const int hi = lane >> 4;
  const int rowbase = blockIdx.x * 16;
  const int colbase = blockIdx.y * 64;

  v8f acc0 = {}, acc1 = {}, acc2 = {}, acc3 = {};

  for (int kb = 0; kb < D_; kb += 32) {
    AFrag a;
    const uint32_t* arow = (const uint32_t*)(A + (size_t)(rowbase + n) * D_ + kb);
    #pragma unroll
    for (int j = 0; j < 4; ++j) {
      int k0 = hi * 8 + 2 * j;
      int k1 = 16 + hi * 8 + 2 * j;
      a.u[j]     = arow[k0 >> 1];
      a.u[j + 4] = arow[k1 >> 1];
    }
    #pragma unroll
    for (int t = 0; t < 4; ++t) {
      AFrag bf;
      const uint32_t* wrow =
          (const uint32_t*)(Wh + (size_t)(colbase + t * 16 + n) * D_ + kb);
      #pragma unroll
      for (int j = 0; j < 8; ++j) bf.u[j] = wrow[(j >> 2) * 8 + hi * 4 + (j & 3)];
      v8f* ac = (t == 0) ? &acc0 : (t == 1) ? &acc1 : (t == 2) ? &acc2 : &acc3;
      *ac = __builtin_amdgcn_wmma_f32_16x16x32_f16(false, a.v, false, bf.v,
                                                   (short)0, *ac, false, false);
    }
  }

  #pragma unroll
  for (int t = 0; t < 4; ++t) {
    const v8f ac = (t == 0) ? acc0 : (t == 1) ? acc1 : (t == 2) ? acc2 : acc3;
    const int col = colbase + t * 16 + n;
    const float bv = bias[col];
    #pragma unroll
    for (int r = 0; r < 8; ++r) {
      const int row = rowbase + r + 8 * hi;
      out[(size_t)row * D_ + col] = ac[r] + bv;
    }
  }
}

// ---------------------------------------------------------------------------
extern "C" void kernel_launch(void* const* d_in, const int* in_sizes, int n_in,
                              void* d_out, int out_size, void* d_ws, size_t ws_size,
                              hipStream_t stream) {
  const float* q    = (const float*)d_in[0];
  const float* k    = (const float*)d_in[1];
  const float* v    = (const float*)d_in[2];
  const int*   mask = (const int*)d_in[3];
  const float* Wq = (const float*)d_in[4];  const float* bq = (const float*)d_in[5];
  const float* Wk = (const float*)d_in[6];  const float* bk = (const float*)d_in[7];
  const float* Wv = (const float*)d_in[8];  const float* bv = (const float*)d_in[9];
  const float* Wo = (const float*)d_in[10]; const float* bo = (const float*)d_in[11];
  float* out = (float*)d_out;

  char* ws = (char*)d_ws;
  const size_t MB = (size_t)1 << 20;
  _Float16* Qb   = (_Float16*)(ws + 0 * MB);
  _Float16* Kb   = (_Float16*)(ws + 16 * MB);
  _Float16* VTb  = (_Float16*)(ws + 32 * MB);
  _Float16* AO   = (_Float16*)(ws + 48 * MB);
  _Float16* Wqh  = (_Float16*)(ws + 64 * MB);
  _Float16* Wkh  = (_Float16*)(ws + 66 * MB);
  _Float16* Wvh  = (_Float16*)(ws + 68 * MB);
  _Float16* Woh  = (_Float16*)(ws + 70 * MB);
  uint32_t* mbit = (uint32_t*)(ws + 72 * MB);

  cvt_f32_f16_kernel<<<dim3(1024), dim3(256), 0, stream>>>(Wq, Wqh, D_ * D_ / 4);
  cvt_f32_f16_kernel<<<dim3(1024), dim3(256), 0, stream>>>(Wk, Wkh, D_ * D_ / 4);
  cvt_f32_f16_kernel<<<dim3(1024), dim3(256), 0, stream>>>(Wv, Wvh, D_ * D_ / 4);
  cvt_f32_f16_kernel<<<dim3(1024), dim3(256), 0, stream>>>(Wo, Woh, D_ * D_ / 4);

  packmask_kernel<<<dim3(512), dim3(256), 0, stream>>>(mask, mbit);

  dim3 pgrid(B_ * S_ / 16, D_ / 64);
  proj_kernel<<<pgrid, dim3(32), 0, stream>>>(q, Wqh, bq, (unsigned short*)Qb, 0);
  proj_kernel<<<pgrid, dim3(32), 0, stream>>>(k, Wkh, bk, (unsigned short*)Kb, 0);
  proj_kernel<<<pgrid, dim3(32), 0, stream>>>(v, Wvh, bv, (unsigned short*)VTb, 1);

  attn_kernel<<<dim3(S_ / (16 * WAVES_), H_, B_), dim3(32 * WAVES_), 0, stream>>>(
      Qb, Kb, VTb, mbit, AO);

  oproj_kernel<<<pgrid, dim3(32), 0, stream>>>(AO, Woh, bo, out);
}
